// TraceSemanticHandshakeV342_2534030705215
// MI455X (gfx1250) — matrix operand order repaired and checked by reading, not compiled, not run on hardware
//
#include <hip/hip_runtime.h>
#include <hip/hip_bf16.h>

typedef __attribute__((ext_vector_type(8)))  _Float16 v8h;
typedef __attribute__((ext_vector_type(16))) _Float16 v16h;
typedef __attribute__((ext_vector_type(8)))  float    v8f;

#define HIDDEN 256
#define CAT    1283
#define CATP   1312      // 41 * 32 (K padded for GEMM1)
#define KK1    41        // k-steps of 32 for GEMM1
#define KK2    8         // k-steps of 32 for GEMM2
#define NTILES 16        // 256 / 16 output-column tiles
#define ROWS   32        // rows (M) per block = two 16-row M tiles

// dynamic LDS partition (bytes)
#define XH_BYTES   (ROWS * CATP * 2)        // 83968
#define Y1_BYTES   (ROWS * HIDDEN * 2)      // 16384
#define RED_BYTES  (ROWS * 8 * 4)           // 1024 (per array)
#define SMEM_BYTES (XH_BYTES + Y1_BYTES + 2 * RED_BYTES + 2 * ROWS * 4)

__device__ __forceinline__ float gelu_erf(float x) {
    // exact erf GELU, matches torch/jax approximate=False
    return 0.5f * x * (1.0f + erff(x * 0.70710678118654752f));
}

// ---------------------------------------------------------------------------
// Pack W1 (CAT x 256 fp32, zero-padded to CATP) and W2 (256 x 256 fp32) into
// f16 WMMA B-fragment order:
//   flat = ((ntile*KK + kk)*32 + lane)*16 + 2*j + h
//   n = ntile*16 + (lane & 15)
//   k = kk*32 + (lane<16 ? 0 : 16) + 2*j + h     (B 32x16, contiguous k order, same as A)
// Each lane's fragment is 16 contiguous halfs (32B) -> single vector load.
// ---------------------------------------------------------------------------
__global__ void pack_weights_kernel(const float* __restrict__ W1,
                                    const float* __restrict__ W2,
                                    _Float16* __restrict__ w1p,
                                    _Float16* __restrict__ w2p) {
    const int N1 = NTILES * KK1 * 32 * 16;   // 335872
    const int N2 = NTILES * KK2 * 32 * 16;   // 65536
    int i = blockIdx.x * 256 + threadIdx.x;
    if (i < N1) {
        int h    = i & 1;
        int j    = (i >> 1) & 7;
        int lane = (i >> 4) & 31;
        int tk   = i >> 9;               // ntile*KK1 + kk
        int kk   = tk % KK1;
        int nt   = tk / KK1;
        int n = nt * 16 + (lane & 15);
        int k = kk * 32 + ((lane < 16) ? 0 : 16) + 2 * j + h;
        w1p[i] = (_Float16)((k < CAT) ? W1[k * HIDDEN + n] : 0.0f);
    } else if (i < N1 + N2) {
        int q    = i - N1;
        int h    = q & 1;
        int j    = (q >> 1) & 7;
        int lane = (q >> 4) & 31;
        int tk   = q >> 9;               // ntile*KK2 + kk
        int kk   = tk % KK2;
        int nt   = tk / KK2;
        int n = nt * 16 + (lane & 15);
        int k = kk * 32 + ((lane < 16) ? 0 : 16) + 2 * j + h;
        w2p[q] = (_Float16)W2[k * HIDDEN + n];
    }
}

// ---------------------------------------------------------------------------
// Fused concat + LayerNorm + (Linear->GELU) x2, one 32-row tile per block.
// 256 threads = 8 wave32; wave w owns column tiles {2w, 2w+1} and both
// 16-row M tiles -> 4 accumulators, 4 WMMA per (2 A-frag + 2 B-frag) loads.
// ---------------------------------------------------------------------------
__global__ void __launch_bounds__(256)
fused_mlp_kernel(const float* __restrict__ point_token,
                 const float* __restrict__ step_hidden,
                 const float* __restrict__ point_dyn,
                 const float* __restrict__ point_sem,
                 const float* __restrict__ pit,
                 const float* __restrict__ rel_pred,
                 const float* __restrict__ vis_prob,
                 const float* __restrict__ gamma,
                 const float* __restrict__ beta,
                 const float* __restrict__ b1,
                 const float* __restrict__ b2,
                 const _Float16* __restrict__ w1p,
                 const _Float16* __restrict__ w2p,
                 float* __restrict__ out) {
    extern __shared__ __align__(16) char smem[];
    _Float16* xh   = (_Float16*)smem;                              // ROWS x CATP
    _Float16* y1   = (_Float16*)(smem + XH_BYTES);                 // ROWS x HIDDEN
    float*    reds = (float*)(smem + XH_BYTES + Y1_BYTES);         // ROWS x 8
    float*    redq = reds + ROWS * 8;                              // ROWS x 8
    float*    mu_s = redq + ROWS * 8;                              // ROWS
    float*    rs_s = mu_s + ROWS;                                  // ROWS

    const int tid  = threadIdx.x;
    const int blk  = blockIdx.x;
    const int lane = tid & 31;
    const int wv   = tid >> 5;

    // ---- Phase 1: gather concat, cast to f16 into LDS (pad cols -> 0) ----
    for (int idx = tid; idx < ROWS * CATP; idx += 256) {
        int row = idx / CATP;
        int c   = idx - row * CATP;
        int gr  = blk * ROWS + row;          // global row in [0, 65536)
        int bi  = gr >> 14;                  // / (256*64)
        int rem = gr & 16383;
        int mi  = rem >> 6;
        int hh  = rem & 63;
        float v = 0.0f;
        if      (c < 256)   v = point_token[(bi * 256 + mi) * 256 + c];
        else if (c < 512)   v = step_hidden[(bi * 64 + hh) * 256 + (c - 256)];
        else if (c < 768)   v = point_dyn[gr * 256 + (c - 512)];
        else if (c < 1024)  v = point_sem[gr * 256 + (c - 768)];
        else if (c < 1280)  v = pit[gr * 256 + (c - 1024)];
        else if (c < 1282)  v = rel_pred[gr * 2 + (c - 1280)];
        else if (c == 1282) v = vis_prob[gr];
        xh[idx] = (_Float16)v;
    }
    __syncthreads();

    // ---- Phase 2: LayerNorm stats, 8 threads per row ----
    {
        int row  = tid >> 3;
        int part = tid & 7;
        float s = 0.0f, q = 0.0f;
        for (int c = part; c < CAT; c += 8) {
            float x = (float)xh[row * CATP + c];
            s += x;
            q += x * x;
        }
        reds[row * 8 + part] = s;
        redq[row * 8 + part] = q;
    }
    __syncthreads();
    if (tid < ROWS) {
        float s = 0.0f, q = 0.0f;
        for (int p = 0; p < 8; ++p) { s += reds[tid * 8 + p]; q += redq[tid * 8 + p]; }
        float mu  = s * (1.0f / (float)CAT);
        float var = q * (1.0f / (float)CAT) - mu * mu;
        mu_s[tid] = mu;
        rs_s[tid] = rsqrtf(var + 1e-5f);
    }
    __syncthreads();

    // ---- Phase 3: affine normalize in place ----
    for (int idx = tid; idx < ROWS * CATP; idx += 256) {
        int row = idx / CATP;
        int c   = idx - row * CATP;
        if (c < CAT) {
            float x = (float)xh[idx];
            x = (x - mu_s[row]) * rs_s[row] * gamma[c] + beta[c];
            xh[idx] = (_Float16)x;
        }
    }
    __syncthreads();

    // ---- Phase 4: GEMM1 (32 x CATP) @ (CATP x 256), f16 WMMA, f32 acc ----
    const int m    = lane & 15;              // A-matrix row within M tile
    const int ksel = (lane < 16) ? 0 : 16;   // A K offset (contiguous k = 16h + s)
    const int nt0  = wv * 2;
    const int nt1  = wv * 2 + 1;

    v8f acc00 = {}; v8f acc01 = {};          // (mtile0, nt0/nt1)
    v8f acc10 = {}; v8f acc11 = {};          // (mtile1, nt0/nt1)
    for (int kk = 0; kk < KK1; ++kk) {
        const _Float16* ar0 = &xh[m * CATP + kk * 32 + ksel];
        const _Float16* ar1 = &xh[(m + 16) * CATP + kk * 32 + ksel];
        v8h a0lo = *(const v8h*)(ar0);
        v8h a0hi = *(const v8h*)(ar0 + 8);
        v8h a1lo = *(const v8h*)(ar1);
        v8h a1hi = *(const v8h*)(ar1 + 8);
        v16h a0 = __builtin_shufflevector(a0lo, a0hi,
                    0,1,2,3,4,5,6,7,8,9,10,11,12,13,14,15);
        v16h a1 = __builtin_shufflevector(a1lo, a1hi,
                    0,1,2,3,4,5,6,7,8,9,10,11,12,13,14,15);
        v16h bf0 = *(const v16h*)&w1p[((nt0 * KK1 + kk) * 32 + lane) * 16];
        v16h bf1 = *(const v16h*)&w1p[((nt1 * KK1 + kk) * 32 + lane) * 16];
        acc00 = __builtin_amdgcn_wmma_f32_16x16x32_f16(false, a0, false, bf0,
                                                       (short)0, acc00, false, false);
        acc01 = __builtin_amdgcn_wmma_f32_16x16x32_f16(false, a0, false, bf1,
                                                       (short)0, acc01, false, false);
        acc10 = __builtin_amdgcn_wmma_f32_16x16x32_f16(false, a1, false, bf0,
                                                       (short)0, acc10, false, false);
        acc11 = __builtin_amdgcn_wmma_f32_16x16x32_f16(false, a1, false, bf1,
                                                       (short)0, acc11, false, false);
    }

    // bias + GELU -> y1 (f16), C/D layout: M = r + (lane<16?0:8), N = lane&15
    {
        int N  = lane & 15;
        int mb = (lane < 16) ? 0 : 8;
        float bb0 = b1[nt0 * 16 + N];
        float bb1 = b1[nt1 * 16 + N];
        for (int r = 0; r < 8; ++r) {
            int M = r + mb;
            y1[M * HIDDEN + nt0 * 16 + N]        = (_Float16)gelu_erf(acc00[r] + bb0);
            y1[M * HIDDEN + nt1 * 16 + N]        = (_Float16)gelu_erf(acc01[r] + bb1);
            y1[(M + 16) * HIDDEN + nt0 * 16 + N] = (_Float16)gelu_erf(acc10[r] + bb0);
            y1[(M + 16) * HIDDEN + nt1 * 16 + N] = (_Float16)gelu_erf(acc11[r] + bb1);
        }
    }
    __syncthreads();

    // ---- Phase 5: GEMM2 (32 x 256) @ (256 x 256) ----
    v8f d00 = {}; v8f d01 = {};
    v8f d10 = {}; v8f d11 = {};
    for (int kk = 0; kk < KK2; ++kk) {
        const _Float16* ar0 = &y1[m * HIDDEN + kk * 32 + ksel];
        const _Float16* ar1 = &y1[(m + 16) * HIDDEN + kk * 32 + ksel];
        v8h a0lo = *(const v8h*)(ar0);
        v8h a0hi = *(const v8h*)(ar0 + 8);
        v8h a1lo = *(const v8h*)(ar1);
        v8h a1hi = *(const v8h*)(ar1 + 8);
        v16h a0 = __builtin_shufflevector(a0lo, a0hi,
                    0,1,2,3,4,5,6,7,8,9,10,11,12,13,14,15);
        v16h a1 = __builtin_shufflevector(a1lo, a1hi,
                    0,1,2,3,4,5,6,7,8,9,10,11,12,13,14,15);
        v16h bf0 = *(const v16h*)&w2p[((nt0 * KK2 + kk) * 32 + lane) * 16];
        v16h bf1 = *(const v16h*)&w2p[((nt1 * KK2 + kk) * 32 + lane) * 16];
        d00 = __builtin_amdgcn_wmma_f32_16x16x32_f16(false, a0, false, bf0,
                                                     (short)0, d00, false, false);
        d01 = __builtin_amdgcn_wmma_f32_16x16x32_f16(false, a0, false, bf1,
                                                     (short)0, d01, false, false);
        d10 = __builtin_amdgcn_wmma_f32_16x16x32_f16(false, a1, false, bf0,
                                                     (short)0, d10, false, false);
        d11 = __builtin_amdgcn_wmma_f32_16x16x32_f16(false, a1, false, bf1,
                                                     (short)0, d11, false, false);
    }

    // bias + GELU -> global fp32 output
    {
        int N  = lane & 15;
        int mb = (lane < 16) ? 0 : 8;
        float bb0 = b2[nt0 * 16 + N];
        float bb1 = b2[nt1 * 16 + N];
        for (int r = 0; r < 8; ++r) {
            int M   = r + mb;
            int gr0 = blk * ROWS + M;
            int gr1 = gr0 + 16;
            out[gr0 * HIDDEN + nt0 * 16 + N] = gelu_erf(d00[r] + bb0);
            out[gr0 * HIDDEN + nt1 * 16 + N] = gelu_erf(d01[r] + bb1);
            out[gr1 * HIDDEN + nt0 * 16 + N] = gelu_erf(d10[r] + bb0);
            out[gr1 * HIDDEN + nt1 * 16 + N] = gelu_erf(d11[r] + bb1);
        }
    }
}

extern "C" void kernel_launch(void* const* d_in, const int* in_sizes, int n_in,
                              void* d_out, int out_size, void* d_ws, size_t ws_size,
                              hipStream_t stream) {
    (void)in_sizes; (void)n_in; (void)out_size; (void)ws_size;
    const float* point_token = (const float*)d_in[0];
    const float* step_hidden = (const float*)d_in[1];
    const float* point_dyn   = (const float*)d_in[2];
    const float* point_sem   = (const float*)d_in[3];
    const float* pit         = (const float*)d_in[4];
    const float* rel_pred    = (const float*)d_in[5];
    const float* vis_prob    = (const float*)d_in[6];
    const float* ln_gamma    = (const float*)d_in[7];
    const float* ln_beta     = (const float*)d_in[8];
    const float* W1          = (const float*)d_in[9];
    const float* b1          = (const float*)d_in[10];
    const float* W2          = (const float*)d_in[11];
    const float* b2          = (const float*)d_in[12];
    float* out = (float*)d_out;

    _Float16* w1p = (_Float16*)d_ws;                       // 335872 halfs (671744 B)
    _Float16* w2p = w1p + NTILES * KK1 * 32 * 16;          // 65536 halfs  (131072 B)

    const int PACK_ELEMS = NTILES * KK1 * 32 * 16 + NTILES * KK2 * 32 * 16; // 401408
    pack_weights_kernel<<<PACK_ELEMS / 256, 256, 0, stream>>>(W1, W2, w1p, w2p);

    const int n_rows = 4 * 256 * 64;                       // 65536
    fused_mlp_kernel<<<n_rows / ROWS, 256, SMEM_BYTES, stream>>>(
        point_token, step_hidden, point_dyn, point_sem, pit,
        rel_pred, vis_prob, ln_gamma, ln_beta, b1, b2, w1p, w2p, out);
}
